// ShiftedWindowAttention_68393059222078
// MI455X (gfx1250) — matrix operand order repaired and checked by reading, not compiled, not run on hardware
//
#include <hip/hip_runtime.h>
#include <hip/hip_bf16.h>

typedef __attribute__((ext_vector_type(16))) _Float16 v16h;
typedef __attribute__((ext_vector_type(8)))  _Float16 v8h;
typedef __attribute__((ext_vector_type(4)))  _Float16 v4h;
typedef __attribute__((ext_vector_type(8)))  float    v8f;
typedef __attribute__((ext_vector_type(4)))  float    v4f;
typedef __attribute__((ext_vector_type(4)))  int      v4i;

#define NHEADS 12
#define DH     64
#define CCH    768
#define LWIN   49     // 7*7 tokens per window
#define NWIN   2048   // 32 * (56/7)*(56/7)
#define MROWS  100352 // NWIN * LWIN
#define SPITCH 68     // padded pitch (floats) of transposed score matrix

#if __has_builtin(__builtin_amdgcn_global_load_async_to_lds_b128)
#define HAVE_ASYNC_LDS 1
#else
#define HAVE_ASYNC_LDS 0
#endif

// 16-byte memory -> LDS copy: async (ASYNCcnt-tracked, no VGPR bounce) when
// available, otherwise a plain VGPR-mediated copy.
static __device__ __forceinline__
void cp_g2l_b128(_Float16* dst_lds, const _Float16* src_glob) {
#if HAVE_ASYNC_LDS
  __builtin_amdgcn_global_load_async_to_lds_b128(
      (v4i*)src_glob, (v4i*)dst_lds, 0, 0);
#else
  *(v8h*)dst_lds = *(const v8h*)src_glob;
#endif
}

static __device__ __forceinline__ void async_wait0() {
#if HAVE_ASYNC_LDS
#if __has_builtin(__builtin_amdgcn_s_wait_asynccnt)
  __builtin_amdgcn_s_wait_asynccnt(0);
#else
  asm volatile("s_wait_asynccnt 0x0" ::: "memory");
#endif
#endif
}

// ---------------------------------------------------------------------------
// Per-lane operand loaders for V_WMMA_F32_16X16X32_F16 (wave32).
// A-matrix 16x32 f16: lanes 0-15 hold row M=lane, elements K0..15 (contiguous);
//                     lanes 16-31 hold row M=lane-16, elements K16..31.
// B-matrix 32x16 f16: lane n<16 holds column n, elements K0..15 (contiguous);
//                     lane n+16 holds column n, elements K16..31.
// C/D f32 16x16: vgpr r -> row r + 8*(lane/16); lane%16 -> column.
// ---------------------------------------------------------------------------
static __device__ __forceinline__
v16h load_a16(const _Float16* __restrict__ base, int pitch, int mrow0, int k0, int lane) {
  const int m    = mrow0 + (lane & 15);
  const int half = lane >> 4;
  const _Float16* p0 = base + m * pitch + k0 + half * 16;
  const _Float16* p1 = base + m * pitch + k0 + 8 + half * 16;
  v8h lo = *(const v8h*)p0;
  v8h hi = *(const v8h*)p1;
  v16h r;
#pragma unroll
  for (int i = 0; i < 8; ++i) { r[i] = lo[i]; r[i + 8] = hi[i]; }
  return r;
}

static __device__ __forceinline__
v16h load_b16(const _Float16* __restrict__ p) {  // 16 contiguous f16 at p
  v8h lo = *(const v8h*)p;
  v8h hi = *(const v8h*)(p + 8);
  v16h r;
#pragma unroll
  for (int i = 0; i < 8; ++i) { r[i] = lo[i]; r[i + 8] = hi[i]; }
  return r;
}

static __device__ __forceinline__
v8f wmma_f16(v16h a, v16h b, v8f c) {
  return __builtin_amdgcn_wmma_f32_16x16x32_f16(false, a, false, b, (short)0, c,
                                                false, false);
}

// ---------------------------------------------------------------------------
// Kernel 0: fp32 -> fp16 weight conversion
// ---------------------------------------------------------------------------
__global__ void cvt_f32_f16(const float* __restrict__ src,
                            _Float16* __restrict__ dst, int n) {
  for (int i = blockIdx.x * blockDim.x + threadIdx.x; i < n;
       i += gridDim.x * blockDim.x)
    dst[i] = (_Float16)src[i];
}

// ---------------------------------------------------------------------------
// Kernel 1: per-(window, head) attention.
// grid = (NWIN, NHEADS), block = 256 (8 waves). Each wave owns 2 of 16
// 16x16 output tiles of the padded 64x64 matrices.
// ---------------------------------------------------------------------------
__global__ void __launch_bounds__(256)
win_attn(const float* __restrict__ x,
         const _Float16* __restrict__ Wq16,
         const _Float16* __restrict__ Wk16,
         const _Float16* __restrict__ Wv16,
         _Float16* __restrict__ Out16) {
  __shared__ __align__(16) _Float16 Ws[3][64 * 64];  // Wq/Wk/Wv head slabs
  __shared__ __align__(16) _Float16 Xs[64 * 64];     // tokens x dh (padded)
  __shared__ __align__(16) _Float16 Qs[64 * 64];     // pre-scaled by 1/8
  __shared__ __align__(16) _Float16 Ks[64 * 64];
  __shared__ __align__(16) _Float16 Vts[64 * 64];    // V transposed: [d][m]
  __shared__ __align__(16) _Float16 Ps[64 * 64];     // softmax probs (row major)
  __shared__ __align__(16) float    St[64 * SPITCH]; // scores transposed: [m][l]

  const int win  = blockIdx.x;
  const int h    = blockIdx.y;
  const int tid  = threadIdx.x;
  const int w    = tid >> 5;
  const int lane = tid & 31;

  // ---- async-stage the three 64x64 f16 weight slabs into LDS ----
  {
    const _Float16* gW[3] = {Wq16 + h * DH * DH, Wk16 + h * DH * DH,
                             Wv16 + h * DH * DH};
    for (int i = tid; i < 3 * 512; i += 256) {   // 512 v8h chunks per matrix
      const int mat = i >> 9;
      const int off = (i & 511) * 8;
      cp_g2l_b128(&Ws[mat][off], gW[mat] + off);
    }
  }

  // ---- stage X (f32 -> f16) into LDS, zero-pad rows 49..63; float4 loads ----
  const int b  = win >> 6;
  const int wy = (win >> 3) & 7;
  const int wx = win & 7;
  for (int idx = tid; idx < 64 * 16; idx += 256) {
    const int row = idx >> 4;
    const int c4  = (idx & 15) * 4;
    v4h hv = {};
    if (row < LWIN) {
      const int iy = row / 7, ix = row - iy * 7;
      const int gy = wy * 7 + iy, gx = wx * 7 + ix;
      v4f f = *(const v4f*)&x[((size_t)(b * 56 + gy) * 56 + gx) * CCH +
                              h * DH + c4];
#pragma unroll
      for (int i = 0; i < 4; ++i) hv[i] = (_Float16)f[i];
    }
    *(v4h*)&Xs[row * 64 + c4] = hv;
  }
  async_wait0();
  __syncthreads();

  // ---- stage 1: Q = (X*Wq^T)/8, K = X*Wk^T, V^T = (X*Wv^T)^T ----
#pragma unroll
  for (int mm = 0; mm < 3; ++mm) {
#pragma unroll
    for (int tt = 0; tt < 2; ++tt) {
      const int t  = w * 2 + tt;
      const int mt = t >> 2, nt = t & 3;
      v8f acc = {};
#pragma unroll
      for (int k0 = 0; k0 < DH; k0 += 32) {
        v16h a = load_a16(Xs, 64, mt * 16, k0, lane);
        // B[k][n] = W[n][k]  (y = x @ W^T)  -> contiguous row reads
        v16h bb = load_b16(&Ws[mm][(nt * 16 + (lane & 15)) * DH + k0 +
                                   (lane >> 4) * 16]);
        acc = wmma_f16(a, bb, acc);
      }
      const int col   = nt * 16 + (lane & 15);
      const int rbase = mt * 16 + (lane >> 4) * 8;
      if (mm == 2) {           // V transposed: contiguous -> one b128 store
        v8h p;
#pragma unroll
        for (int r = 0; r < 8; ++r) p[r] = (_Float16)acc[r];
        *(v8h*)&Vts[col * 64 + rbase] = p;
      } else if (mm == 0) {    // Q, pre-scaled by 1/sqrt(64)
#pragma unroll
        for (int r = 0; r < 8; ++r)
          Qs[(rbase + r) * 64 + col] = (_Float16)(acc[r] * 0.125f);
      } else {                 // K row-major (stage-2 B operand wants rows)
#pragma unroll
        for (int r = 0; r < 8; ++r)
          Ks[(rbase + r) * 64 + col] = (_Float16)acc[r];
      }
    }
  }
  __syncthreads();

  // ---- stage 2: S^T = (Q K^T)^T, stored column-major (contiguous b128) ----
#pragma unroll
  for (int tt = 0; tt < 2; ++tt) {
    const int t  = w * 2 + tt;
    const int mt = t >> 2, nt = t & 3;
    v8f acc = {};
#pragma unroll
    for (int k0 = 0; k0 < DH; k0 += 32) {
      v16h a = load_a16(Qs, 64, mt * 16, k0, lane);
      // B[e][m] = K[m][e] -> contiguous reads of K rows
      v16h bb = load_b16(Ks + (nt * 16 + (lane & 15)) * 64 + k0 +
                         (lane >> 4) * 16);
      acc = wmma_f16(a, bb, acc);
    }
    const int col   = nt * 16 + (lane & 15);       // score column m
    const int rbase = mt * 16 + (lane >> 4) * 8;   // score row l
    v4f lo, hi;
#pragma unroll
    for (int i = 0; i < 4; ++i) { lo[i] = acc[i]; hi[i] = acc[i + 4]; }
    *(v4f*)&St[col * SPITCH + rbase]     = lo;
    *(v4f*)&St[col * SPITCH + rbase + 4] = hi;
  }
  __syncthreads();

  // ---- stage 3: masked softmax; 4 threads per row, quad shuffle reduce ----
  {
    const int l = tid >> 2;     // row (token)
    const int q = tid & 3;      // quad member
    if (l < LWIN) {
      float mx = -3.4e38f;
      for (int c = q; c < LWIN; c += 4) mx = fmaxf(mx, St[c * SPITCH + l]);
      mx = fmaxf(mx, __shfl_xor(mx, 1, 32));
      mx = fmaxf(mx, __shfl_xor(mx, 2, 32));
      float sum = 0.0f;
      for (int c = q; c < LWIN; c += 4) {
        float e = __expf(St[c * SPITCH + l] - mx);
        St[c * SPITCH + l] = e;
        sum += e;
      }
      sum += __shfl_xor(sum, 1, 32);
      sum += __shfl_xor(sum, 2, 32);
      const float inv = 1.0f / sum;
      for (int c = q; c < 64; c += 4)
        Ps[l * 64 + c] = (c < LWIN) ? (_Float16)(St[c * SPITCH + l] * inv)
                                    : (_Float16)0.0f;
    } else {
      for (int c = q; c < 64; c += 4) Ps[l * 64 + c] = (_Float16)0.0f;
    }
  }
  __syncthreads();

  // ---- stage 4: O = P * V, store f16 to concat-head workspace ----
#pragma unroll
  for (int tt = 0; tt < 2; ++tt) {
    const int t  = w * 2 + tt;
    const int mt = t >> 2, nt = t & 3;
    v8f acc = {};
#pragma unroll
    for (int k0 = 0; k0 < 64; k0 += 32) {
      v16h a = load_a16(Ps, 64, mt * 16, k0, lane);
      // B[m][d] = V[m][d] = Vts[d][m] -> contiguous reads of Vts rows
      v16h bb = load_b16(Vts + (nt * 16 + (lane & 15)) * 64 + k0 +
                         (lane >> 4) * 16);
      acc = wmma_f16(a, bb, acc);
    }
    const int col   = h * DH + nt * 16 + (lane & 15);
    const int rbase = mt * 16 + (lane >> 4) * 8;
#pragma unroll
    for (int r = 0; r < 8; ++r) {
      const int row = rbase + r;
      if (row < LWIN)
        Out16[(size_t)(win * LWIN + row) * CCH + col] = (_Float16)acc[r];
    }
  }
}

// ---------------------------------------------------------------------------
// Kernel 2: Y = Out * Wo^T + bo.  M=100352, N=768, K=768.
// grid = (1568, 6); each block makes a 64x128 output tile; 8 waves x 4 tiles
// sharing one A operand per k-step. A is double-buffered through LDS with
// async memory->LDS copies overlapping WMMA on the previous slab.
// ---------------------------------------------------------------------------
__global__ void __launch_bounds__(256)
out_gemm(const _Float16* __restrict__ Out16,
         const _Float16* __restrict__ Wo16,
         const float* __restrict__ bo,
         float* __restrict__ y) {
  __shared__ __align__(16) _Float16 As[2][64 * 32];

  const int tid  = threadIdx.x;
  const int w    = tid >> 5;
  const int lane = tid & 31;
  const size_t m0 = (size_t)blockIdx.x * 64;
  const int    n0 = blockIdx.y * 128;

  const int mt = w >> 1;          // row tile (shared by wave pair)
  const int nb = (w & 1) * 4;     // first of 4 column tiles for this wave

  const int arow = tid >> 2;      // cooperative A-copy mapping (one b128 each)
  const int aseg = tid & 3;
  const _Float16* asrc = Out16 + (m0 + arow) * CCH + aseg * 8;
  _Float16* adst0 = &As[0][arow * 32 + aseg * 8];
  _Float16* adst1 = &As[1][arow * 32 + aseg * 8];

  v8f acc[4] = {{}, {}, {}, {}};

  cp_g2l_b128(adst0, asrc);       // prologue: k0 = 0 into buffer 0

  for (int k0 = 0; k0 < CCH; k0 += 32) {
    const int buf = (k0 >> 5) & 1;
    async_wait0();                // my slab (issued last iter) is in LDS
    __syncthreads();              // everyone's slab is in LDS; prev reads done
    if (k0 + 32 < CCH)            // overlap next slab fetch with WMMA
      cp_g2l_b128(buf ? adst0 : adst1, asrc + k0 + 32);
    v16h a = load_a16(As[buf], 32, mt * 16, 0, lane);
#pragma unroll
    for (int j = 0; j < 4; ++j) {
      const int nt = nb + j;
      // B[c][o] = Wo[o][c] -> contiguous reads of Wo rows (L2-resident)
      v16h bb = load_b16(Wo16 + (size_t)(n0 + nt * 16 + (lane & 15)) * CCH +
                         k0 + (lane >> 4) * 16);
      acc[j] = wmma_f16(a, bb, acc[j]);
    }
  }

#pragma unroll
  for (int j = 0; j < 4; ++j) {
    const int col   = n0 + (nb + j) * 16 + (lane & 15);
    const int rbase = mt * 16 + (lane >> 4) * 8;
    const float bias = bo[col];
#pragma unroll
    for (int r = 0; r < 8; ++r)
      y[(m0 + rbase + r) * CCH + col] = acc[j][r] + bias;
  }
}

// ---------------------------------------------------------------------------
extern "C" void kernel_launch(void* const* d_in, const int* in_sizes, int n_in,
                              void* d_out, int out_size, void* d_ws, size_t ws_size,
                              hipStream_t stream) {
  (void)in_sizes; (void)n_in; (void)out_size; (void)ws_size;
  const float* x  = (const float*)d_in[0];
  const float* Wq = (const float*)d_in[1];
  const float* Wk = (const float*)d_in[2];
  const float* Wv = (const float*)d_in[3];
  const float* Wo = (const float*)d_in[4];
  const float* bo = (const float*)d_in[5];

  _Float16* Wq16  = (_Float16*)d_ws;
  _Float16* Wk16  = Wq16 + NHEADS * DH * DH;
  _Float16* Wv16  = Wk16 + NHEADS * DH * DH;
  _Float16* Wo16  = Wv16 + NHEADS * DH * DH;
  _Float16* Out16 = Wo16 + CCH * CCH;   // 100352 x 768 f16

  cvt_f32_f16<<<48, 256, 0, stream>>>(Wq, Wq16, NHEADS * DH * DH);
  cvt_f32_f16<<<48, 256, 0, stream>>>(Wk, Wk16, NHEADS * DH * DH);
  cvt_f32_f16<<<48, 256, 0, stream>>>(Wv, Wv16, NHEADS * DH * DH);
  cvt_f32_f16<<<192, 256, 0, stream>>>(Wo, Wo16, CCH * CCH);

  win_attn<<<dim3(NWIN, NHEADS), 256, 0, stream>>>(x, Wq16, Wk16, Wv16, Out16);

  out_gemm<<<dim3(MROWS / 64, CCH / 128), 256, 0, stream>>>(Out16, Wo16, bo,
                                                            (float*)d_out);
}
